// DynamicTokenBlock_72464688218761
// MI455X (gfx1250) — matrix operand order repaired and checked by reading, not compiled, not run on hardware
//
#include <hip/hip_runtime.h>
#include <hip/hip_bf16.h>

#ifndef USE_ASYNC_LDS
#define USE_ASYNC_LDS 1     // gfx1250 GLOBAL_LOAD_ASYNC_TO_LDS_B128 pipeline
#endif

// ---------------------------------------------------------------------------
// Types / helpers
// ---------------------------------------------------------------------------
typedef __attribute__((ext_vector_type(16))) __bf16 v16bf;
typedef __attribute__((ext_vector_type(8)))  float  v8f;

union BfFrag {           // 32-byte WMMA operand as two 16B chunks
    uint4 q[2];
    v16bf v;
};

__device__ __forceinline__ __bf16 f2bf(float x) {
    unsigned u = __builtin_bit_cast(unsigned, x);
    unsigned r = (u + 0x7FFFu + ((u >> 16) & 1u)) >> 16;
    unsigned short s = (unsigned short)r;
    return __builtin_bit_cast(__bf16, s);
}

__device__ __forceinline__ float gelu_exact(float x) {
    return 0.5f * x * (1.0f + erff(x * 0.70710678118654752f));
}

#if USE_ASYNC_LDS
// GVS-mode async copy: LDS[lds_off] = MEM[sbase + voff], 16 bytes, ASYNCcnt.
__device__ __forceinline__ void async_ld_b128(unsigned lds_off, unsigned voff,
                                              const void* sbase) {
    unsigned long long base = (unsigned long long)(uintptr_t)sbase;
    asm volatile("global_load_async_to_lds_b128 %0, %1, %2 offset:0"
                 :: "v"(lds_off), "v"(voff), "s"(base) : "memory");
}
__device__ __forceinline__ void wait_async0() {
    asm volatile("s_wait_asynccnt 0x0" ::: "memory");
}
#endif

// ---------------------------------------------------------------------------
// Transposing f32 -> bf16 cast: in [bat, R, C] row-major -> out [bat, C, R].
// ---------------------------------------------------------------------------
__global__ void cast_bf16_T_kernel(const float* __restrict__ in,
                                   __bf16* __restrict__ out,
                                   int R, int C, long long total) {
    long long i = (long long)blockIdx.x * blockDim.x + threadIdx.x;
    long long stride = (long long)gridDim.x * blockDim.x;
    long long rc = (long long)R * C;
    for (; i < total; i += stride) {
        long long bat = i / rc, rem = i - bat * rc;
        long long r = rem / C, c = rem - r * C;
        out[bat * rc + c * (long long)R + r] = f2bf(in[i]);
    }
}

// ---------------------------------------------------------------------------
// Generic LayerNorm (+optional second strided input) -> bf16.
// Row r decomposes as (r/S, r%S); element i:
//   in0[(r/S)*P0 + (r%S)*Q0 + i*ST0]  (+ in1[...] if in1)
// out[(r/S)*OP + (r%S)*OQ + i]
// ---------------------------------------------------------------------------
__global__ void ln_bf16_kernel(const float* __restrict__ in0,
                               long long P0, long long Q0, long long ST0,
                               const float* __restrict__ in1,
                               long long P1, long long Q1, long long ST1,
                               const float* __restrict__ g,
                               const float* __restrict__ bb,
                               __bf16* __restrict__ out,
                               long long OP, long long OQ,
                               int S, int C) {
    int r = blockIdx.x;
    long long hi = r / S, lo = r % S;
    const float* p0 = in0 + hi * P0 + lo * Q0;
    const float* p1 = in1 ? (in1 + hi * P1 + lo * Q1) : nullptr;
    __bf16* po = out + hi * OP + lo * OQ;

    __shared__ float rs[256];
    __shared__ float rs2[256];
    float s = 0.f, s2 = 0.f;
    for (int i = threadIdx.x; i < C; i += 256) {
        float v = p0[(long long)i * ST0];
        if (p1) v += p1[(long long)i * ST1];
        s += v; s2 += v * v;
    }
    rs[threadIdx.x] = s; rs2[threadIdx.x] = s2;
    __syncthreads();
    for (int off = 128; off > 0; off >>= 1) {
        if (threadIdx.x < off) {
            rs[threadIdx.x]  += rs[threadIdx.x + off];
            rs2[threadIdx.x] += rs2[threadIdx.x + off];
        }
        __syncthreads();
    }
    float mean = rs[0] / (float)C;
    float var  = rs2[0] / (float)C - mean * mean;
    float inv  = rsqrtf(var + 1e-5f);
    for (int i = threadIdx.x; i < C; i += 256) {
        float v = p0[(long long)i * ST0];
        if (p1) v += p1[(long long)i * ST1];
        po[i] = f2bf((v - mean) * inv * g[i] + bb[i]);
    }
}

// ---------------------------------------------------------------------------
// Batched GEMM, bf16 x bf16 -> f32 via V_WMMA_F32_16X16X32_BF16.
//   A : row-major [M x Kd], leading dim a_rs (contiguous K by contract)
//   BT: K-major B-transposed [N x Kd], leading dim ldbt
//   C : f32 row-major ldc; optional bias + gelu epilogue.
// Block 256 thr = 8 wave32; macro-tile 128x128, BK=32; wave = 64x32 = 4x2
// WMMA tiles (8 v_wmma / K-step). Double-buffered LDS; next tile staged with
// GLOBAL_LOAD_ASYNC_TO_LDS_B128 while WMMAs run on the current tile; one
// s_wait_asynccnt + one barrier per K-step. LDS rows padded to 40 elems
// (80B): 16B-aligned b128 rows, 20-bank stride => conflict-free fragment
// reads. M,Kd multiples of 128/32. N edge: BT row index clamped (out-of-
// range rows only feed cols >= N which the store guard drops).
// ---------------------------------------------------------------------------
__global__ __launch_bounds__(256)
void gemm_bf16_wmma_kernel(const __bf16* __restrict__ A,
                           long long a_rs, long long a_bs,
                           const __bf16* __restrict__ BT,
                           int ldbt, long long b_bs,
                           float* __restrict__ C, int ldc, long long c_bs,
                           const float* __restrict__ bias, int act,
                           int N, int Kd) {
    __shared__ __align__(16) __bf16 sA [2][128][40];
    __shared__ __align__(16) __bf16 sBt[2][128][40];

    const int m0 = blockIdx.y * 128;
    const int n0 = blockIdx.x * 128;
    A  += (long long)blockIdx.z * a_bs;
    BT += (long long)blockIdx.z * b_bs;
    C  += (long long)blockIdx.z * c_bs;

    const int t    = threadIdx.x;
    const int lane = t & 31;
    const int wave = t >> 5;
    const int wm = (wave & 1) * 64;
    const int wn = (wave >> 1) * 32;

    v8f acc[4][2];
#pragma unroll
    for (int i = 0; i < 4; ++i)
#pragma unroll
        for (int j = 0; j < 2; ++j) acc[i][j] = (v8f){0,0,0,0,0,0,0,0};

    // staging geometry: 2 threads/row, 16 elems (32B) each
    const int row = t >> 1;
    const int ks  = (t & 1) * 16;
    const int gn  = n0 + row;
    const int gnc = (gn < N) ? gn : (N - 1);          // branchless clamp
    const unsigned aoff0 = (unsigned)((((long long)(m0 + row)) * a_rs + ks) * 2);
    const unsigned boff0 = (unsigned)((((long long)gnc) * ldbt + ks) * 2);
#if USE_ASYNC_LDS
    unsigned ldsA0 = (unsigned)(uintptr_t)&sA [0][row][ks];
    unsigned ldsA1 = (unsigned)(uintptr_t)&sA [1][row][ks];
    unsigned ldsB0 = (unsigned)(uintptr_t)&sBt[0][row][ks];
    unsigned ldsB1 = (unsigned)(uintptr_t)&sBt[1][row][ks];
#endif

    auto stage = [&](int buf, int k0) {
#if USE_ASYNC_LDS
        unsigned av = aoff0 + (unsigned)(k0 * 2);
        unsigned bv = boff0 + (unsigned)(k0 * 2);
        unsigned la = buf ? ldsA1 : ldsA0;
        unsigned lb = buf ? ldsB1 : ldsB0;
        async_ld_b128(la,      av,      A);
        async_ld_b128(la + 16, av + 16, A);
        async_ld_b128(lb,      bv,      BT);
        async_ld_b128(lb + 16, bv + 16, BT);
#else
        const uint4* ap = reinterpret_cast<const uint4*>(
            A + (long long)(m0 + row) * a_rs + (k0 + ks));
        const uint4* bp = reinterpret_cast<const uint4*>(
            BT + (long long)gnc * ldbt + (k0 + ks));
        uint4 a0 = ap[0], a1 = ap[1], q0 = bp[0], q1 = bp[1];
        *reinterpret_cast<uint4*>(&sA [buf][row][ks])     = a0;
        *reinterpret_cast<uint4*>(&sA [buf][row][ks + 8]) = a1;
        *reinterpret_cast<uint4*>(&sBt[buf][row][ks])     = q0;
        *reinterpret_cast<uint4*>(&sBt[buf][row][ks + 8]) = q1;
#endif
    };

    stage(0, 0);
#if USE_ASYNC_LDS
    wait_async0();
#endif
    __syncthreads();

    const int lrow   = lane & 15;
    const int khalfA = (lane >> 4) * 8;    // hi lanes: K 8..15 / 24..31
    const int khalfB = (lane >> 4) * 8;    // hi lanes: K 8..15 / 24..31

    int cur = 0;
    for (int k0 = 0; k0 < Kd; k0 += 32) {
        if (k0 + 32 < Kd) stage(cur ^ 1, k0 + 32);   // overlap with WMMAs

        BfFrag afr[4];
#pragma unroll
        for (int tm = 0; tm < 4; ++tm) {
            int r = wm + tm * 16 + lrow;
            afr[tm].q[0] = *reinterpret_cast<const uint4*>(&sA[cur][r][khalfA]);
            afr[tm].q[1] = *reinterpret_cast<const uint4*>(&sA[cur][r][16 + khalfA]);
        }
        BfFrag bfr[2];
#pragma unroll
        for (int tn = 0; tn < 2; ++tn) {
            int c = wn + tn * 16 + lrow;
            bfr[tn].q[0] = *reinterpret_cast<const uint4*>(&sBt[cur][c][khalfB]);
            bfr[tn].q[1] = *reinterpret_cast<const uint4*>(&sBt[cur][c][16 + khalfB]);
        }

#pragma unroll
        for (int tm = 0; tm < 4; ++tm)
#pragma unroll
            for (int tn = 0; tn < 2; ++tn)
                acc[tm][tn] = __builtin_amdgcn_wmma_f32_16x16x32_bf16(
                    false, afr[tm].v, false, bfr[tn].v,
                    (short)0, acc[tm][tn], false, false);

#if USE_ASYNC_LDS
        wait_async0();
#endif
        __syncthreads();          // stores went to the other buffer: 1 barrier
        cur ^= 1;
    }

    // ---- epilogue: bias + optional gelu (C layout per ISA 16x16 f32)
#pragma unroll
    for (int tm = 0; tm < 4; ++tm) {
#pragma unroll
        for (int tn = 0; tn < 2; ++tn) {
            int rbase = m0 + wm + tm * 16 + ((lane >> 4) * 8);
            int col   = n0 + wn + tn * 16 + (lane & 15);
            if (col < N) {
                float bv = bias ? bias[col] : 0.f;
#pragma unroll
                for (int v = 0; v < 8; ++v) {
                    float x = acc[tm][tn][v] + bv;
                    if (act) x = gelu_exact(x);
                    C[(long long)(rbase + v) * ldc + col] = x;
                }
            }
        }
    }
}

// ---------------------------------------------------------------------------
// Split h[65536,513] -> score[65536], x_proj[65536,512]
// ---------------------------------------------------------------------------
__global__ void split_h_kernel(const float* __restrict__ h,
                               float* __restrict__ score,
                               float* __restrict__ xp) {
    long long r = blockIdx.x;
    const float* hr = h + r * 513;
    if (threadIdx.x == 0) score[r] = hr[0];
    for (int i = threadIdx.x; i < 512; i += 256) xp[r * 512 + i] = hr[1 + i];
}

// ---------------------------------------------------------------------------
// Differentiable top-k: 128 sequential softmax rounds per batch row.
// Writes Wmat in BOTH layouts so every downstream GEMM reads contiguous K.
// ---------------------------------------------------------------------------
__global__ __launch_bounds__(1024)
void topk_kernel(const float* __restrict__ score,
                 const float* __restrict__ tauP,
                 __bf16* __restrict__ wkn,     // [B,128,2048]
                 __bf16* __restrict__ wnk) {   // [B,2048,128]
    const int b   = blockIdx.x;
    const int tid = threadIdx.x;
    const float* s = score + (long long)b * 2048;
    __bf16* w  = wkn + (long long)b * 128 * 2048;
    __bf16* wt = wnk + (long long)b * 2048 * 128;

    float tau  = fminf(fmaxf(tauP[0], -2.f), 5.f);
    float tsig = 1.f / (1.f + __expf(-tau));
    float invt = 1.f / tsig;

    float c0 = s[tid], c1 = s[tid + 1024];
    __shared__ float red[1024];

    for (int k = 0; k < 128; ++k) {
        red[tid] = fmaxf(c0, c1);
        __syncthreads();
        for (int off = 512; off > 0; off >>= 1) {
            if (tid < off) red[tid] = fmaxf(red[tid], red[tid + off]);
            __syncthreads();
        }
        float m = red[0];
        __syncthreads();
        float e0 = __expf((c0 - m) * invt);
        float e1 = __expf((c1 - m) * invt);
        red[tid] = e0 + e1;
        __syncthreads();
        for (int off = 512; off > 0; off >>= 1) {
            if (tid < off) red[tid] += red[tid + off];
            __syncthreads();
        }
        float inv = 1.f / red[0];
        __syncthreads();
        float a0 = e0 * inv, a1 = e1 * inv;
        __bf16 h0 = f2bf(a0), h1 = f2bf(a1);
        w[(long long)k * 2048 + tid]          = h0;
        w[(long long)k * 2048 + tid + 1024]   = h1;
        wt[(long long)tid * 128 + k]          = h0;
        wt[(long long)(tid + 1024) * 128 + k] = h1;
        c0 += __logf(fminf(fmaxf(1.f - a0, 1e-6f), 1.f));
        c1 += __logf(fminf(fmaxf(1.f - a1, 1e-6f), 1.f));
    }
}

// ---------------------------------------------------------------------------
// Launch
// ---------------------------------------------------------------------------
extern "C" void kernel_launch(void* const* d_in, const int* in_sizes, int n_in,
                              void* d_out, int out_size, void* d_ws, size_t ws_size,
                              hipStream_t stream) {
    (void)in_sizes; (void)n_in; (void)out_size; (void)ws_size;
    const int Bb = 32, SEQ = 2048, DIM = 512, D = 512;
    const long long NS = (long long)Bb * SEQ;   // 65536

    const float* x     = (const float*)d_in[0];
    const float* tau   = (const float*)d_in[1];
    const float* ln1_g = (const float*)d_in[2];
    const float* ln1_b = (const float*)d_in[3];
    const float* W1    = (const float*)d_in[4];
    const float* b1    = (const float*)d_in[5];
    const float* ln2_g = (const float*)d_in[6];
    const float* ln2_b = (const float*)d_in[7];
    const float* W2    = (const float*)d_in[8];
    const float* b2    = (const float*)d_in[9];
    const float* ln3_g = (const float*)d_in[10];
    const float* ln3_b = (const float*)d_in[11];
    const float* W3    = (const float*)d_in[12];
    const float* b3    = (const float*)d_in[13];
    const float* ln4_g = (const float*)d_in[14];
    const float* ln4_b = (const float*)d_in[15];
    const float* W4    = (const float*)d_in[16];
    const float* b4    = (const float*)d_in[17];
    float* out = (float*)d_out;

    // workspace layout (~512 MB, sequentially reused)
    char* ws = (char*)d_ws;
    const size_t MB = 1024ull * 1024ull;
    __bf16* ABUF  = (__bf16*)(ws + 0);          // 64MB: A1 / xtln / A3 / xprojT / A4
    float*  XTOK  = (float*) (ws + 64  * MB);   // 128MB
    float*  CH    = (float*) (ws + 192 * MB);   // 128MB (later x_proj f32)
    float*  H     = (float*) (ws + 320 * MB);   // 136MB (later Y f32)
    float*  SCORE = (float*) (ws + 456 * MB);   // 1MB
    __bf16* WMB   = (__bf16*)(ws + 457 * MB);   // 16MB Wmat [B,128,2048]
    float*  Zf    = (float*) (ws + 473 * MB);   // 8MB
    __bf16* ZT    = (__bf16*)(ws + 481 * MB);   // 4MB Z^T [B,512,128]
    __bf16* W1T   = (__bf16*)(ws + 485 * MB);   // 0.5MB
    __bf16* W2T   = (__bf16*)(ws + 486 * MB);   // 8MB
    __bf16* W3T   = (__bf16*)(ws + 494 * MB);   // ~0.52MB [513,512]
    __bf16* W4T   = (__bf16*)(ws + 495 * MB);   // 0.5MB
    __bf16* WMT   = (__bf16*)(ws + 496 * MB);   // 16MB Wmat^T [B,2048,128]

    // --- weights -> bf16, K-major (B-operand layout)
    cast_bf16_T_kernel<<<2048, 256, 0, stream>>>(W1, W1T, DIM, D,  (long long)DIM * D);
    cast_bf16_T_kernel<<<4096, 256, 0, stream>>>(W2, W2T, SEQ, SEQ,(long long)SEQ * SEQ);
    cast_bf16_T_kernel<<<2048, 256, 0, stream>>>(W3, W3T, D, D + 1,(long long)D * (D + 1));
    cast_bf16_T_kernel<<<2048, 256, 0, stream>>>(W4, W4T, D, DIM,  (long long)D * DIM);

    // --- ln1(x) -> A1 bf16 [NS, DIM]
    ln_bf16_kernel<<<(int)NS, 256, 0, stream>>>(
        x, 0, DIM, 1, nullptr, 0, 0, 0, ln1_g, ln1_b,
        ABUF, 0, DIM, (int)NS, DIM);

    // --- GEMM1: x_tok = gelu(A1 @ W1 + b1)   [NS,512]
    gemm_bf16_wmma_kernel<<<dim3(4, 512, 1), 256, 0, stream>>>(
        ABUF, DIM, 0, W1T, DIM, 0, XTOK, D, 0, b1, 1, D, DIM);

    // --- ln2 over seq dim (transposed gather) -> xtln bf16 [B,512,2048]
    ln_bf16_kernel<<<Bb * D, 256, 0, stream>>>(
        XTOK, (long long)SEQ * D, 1, D, nullptr, 0, 0, 0, ln2_g, ln2_b,
        ABUF, (long long)D * SEQ, SEQ, D, SEQ);

    // --- GEMM2 (batched 32): ch[b,d,m] = gelu(xtln @ W2 + b2)
    gemm_bf16_wmma_kernel<<<dim3(16, 4, 32), 256, 0, stream>>>(
        ABUF, SEQ, (long long)D * SEQ, W2T, SEQ, 0,
        CH, SEQ, (long long)D * SEQ, b2, 1, SEQ, SEQ);

    // --- ln3(x_tok + ch^T) -> A3 bf16 [NS,512]
    ln_bf16_kernel<<<(int)NS, 256, 0, stream>>>(
        XTOK, (long long)SEQ * D, D, 1,
        CH,   (long long)D * SEQ, 1, SEQ,
        ln3_g, ln3_b, ABUF, (long long)SEQ * D, D, SEQ, D);

    // --- GEMM3: h = A3 @ W3 + b3   [NS,513] (N edge via clamp+store guard)
    gemm_bf16_wmma_kernel<<<dim3(5, 512, 1), 256, 0, stream>>>(
        ABUF, D, 0, W3T, D, 0, H, D + 1, 0, b3, 0, D + 1, D);

    // --- split h -> score, x_proj f32
    split_h_kernel<<<(int)NS, 256, 0, stream>>>(H, SCORE, CH);

    // --- differentiable top-k -> Wmat bf16 (both layouts)
    topk_kernel<<<Bb, 1024, 0, stream>>>(SCORE, tau, WMB, WMT);

    // --- x_proj f32 -> bf16 transposed: xprojT [B,512,2048] (B-operand of Z)
    cast_bf16_T_kernel<<<8192, 256, 0, stream>>>(CH, ABUF, SEQ, D, NS * D);

    // --- Z = Wmat @ x_proj   [B,128,512]
    gemm_bf16_wmma_kernel<<<dim3(4, 1, 32), 256, 0, stream>>>(
        WMB, SEQ, (long long)128 * SEQ,
        ABUF, SEQ, (long long)D * SEQ,
        Zf, D, (long long)128 * D, nullptr, 0, D, SEQ);

    // --- Z -> bf16 transposed: ZT [B,512,128] (B-operand of Y)
    cast_bf16_T_kernel<<<2048, 256, 0, stream>>>(Zf, ZT, 128, D, (long long)Bb * 128 * D);

    // --- Y = Wmat^T @ Z   [B,2048,512]
    gemm_bf16_wmma_kernel<<<dim3(4, 16, 32), 256, 0, stream>>>(
        WMT, 128, (long long)SEQ * 128,
        ZT, 128, (long long)D * 128,
        H, D, (long long)SEQ * D, nullptr, 0, D, 128);

    // --- ln4(x_proj + Y) -> A4 bf16
    ln_bf16_kernel<<<(int)NS, 256, 0, stream>>>(
        CH, 0, D, 1, H, 0, D, 1, ln4_g, ln4_b,
        ABUF, 0, D, (int)NS, D);

    // --- GEMM4: out = A4 @ W4 + b4   [NS,512]
    gemm_bf16_wmma_kernel<<<dim3(4, 512, 1), 256, 0, stream>>>(
        ABUF, D, 0, W4T, D, 0, out, DIM, 0, b4, 0, DIM, D);
}
